// VariationalLSTM_53618371723785
// MI455X (gfx1250) — matrix operand order repaired and checked by reading, not compiled, not run on hardware
//
#include <hip/hip_runtime.h>

typedef __attribute__((ext_vector_type(16))) _Float16 v16h;
typedef __attribute__((ext_vector_type(8)))  _Float16 v8h;
typedef __attribute__((ext_vector_type(8)))  float    v8f;

#define T_STEPS 24
#define HDIM 64
#define NW 8                 // waves per block (wave32)
#define WSZ (256 * 64)       // halves per weight matrix [4H][H]

// Fast sigmoid/tanh: v_exp_f32 + v_rcp_f32 (TRANS ops co-execute with WMMA),
// avoiding clang's ~10-instruction IEEE divide expansion.
__device__ __forceinline__ float sigf(float x) {
    return __builtin_amdgcn_rcpf(1.0f + __expf(-x));
}
__device__ __forceinline__ float tanhfast(float x) {
    float e = __expf(-2.0f * x);
    return (1.0f - e) * __builtin_amdgcn_rcpf(1.0f + e);
}
// DS ops from one wave are processed in order on CDNA5; this mainly stops the
// compiler from reordering cross-lane LDS RAW (staging write -> fragment read).
__device__ __forceinline__ void lds_fence() {
    asm volatile("s_wait_dscnt 0" ::: "memory");
}

// A fragment (16x32, f16): lane<16 holds row=lane, K = k0+[0..7], k0+[16..23];
// lane>=16 holds row=lane-16, K = k0+[8..15], k0+[24..31].  Tile is row-major
// [16][64] f16 in LDS.
__device__ __forceinline__ v16h ldsA(const _Float16* tb, int lane, int k0) {
    const int row = lane & 15, hi = lane >> 4;
    const _Float16* p = tb + row * HDIM + k0 + hi * 8;
    v8h lo = *(const v8h*)p;
    v8h hh = *(const v8h*)(p + 16);
    v16h r;
#pragma unroll
    for (int i = 0; i < 8; ++i) { r[i] = lo[i]; r[i + 8] = hh[i]; }
    return r;
}
// B fragment (32x16, f16): lane<16 holds col=lane K=k0+[0..7], k0+[16..23]; lane>=16 holds
// col=lane-16 K=k0+[8..15], k0+[24..31].  Weights stored row-major [N=256][K=64] f16, so
// B[k][n] = W[n][k] and each lane reads two 8-half runs, 16 apart, of one W row.
__device__ __forceinline__ v16h ldsB(const _Float16* tb, int lane, int k0) {
    const int col = lane & 15, hi = lane >> 4;
    const _Float16* p = tb + col * HDIM + k0 + hi * 8;
    v8h lo = *(const v8h*)p;
    v8h hh = *(const v8h*)(p + 16);
    v16h r;
#pragma unroll
    for (int i = 0; i < 8; ++i) { r[i] = lo[i]; r[i + 8] = hh[i]; }
    return r;
}
__device__ __forceinline__ v8f wmma(v16h a, v16h b, v8f c) {
    return __builtin_amdgcn_wmma_f32_16x16x32_f16(false, a, false, b,
                                                  (short)0, c, false, false);
}

__global__ __launch_bounds__(256) void vlstm_kernel(
    const float* __restrict__ x,
    const float* __restrict__ W_ih1, const float* __restrict__ W_hh1,
    const float* __restrict__ b_ih1, const float* __restrict__ b_hh1,
    const float* __restrict__ W_ih2, const float* __restrict__ W_hh2,
    const float* __restrict__ b_ih2, const float* __restrict__ b_hh2,
    const float* __restrict__ W_out, const float* __restrict__ b_out,
    const float* __restrict__ h1_0, const float* __restrict__ c1_0,
    const float* __restrict__ h2_0, const float* __restrict__ c2_0,
    const float* __restrict__ m_time1, const float* __restrict__ m_time2,
    const float* __restrict__ m_layer1, const float* __restrict__ m_layer2,
    float* __restrict__ out)
{
    __shared__ __align__(16) _Float16 sW[3 * WSZ];        // W_hh1 | W_ih2 | W_hh2 (f16)
    __shared__ __align__(16) float    sB1[256];           // b_ih1 + b_hh1
    __shared__ __align__(16) float    sB2[256];           // b_ih2 + b_hh2
    __shared__ __align__(16) float    sWih1[256];         // W_ih1 (input dim 1)
    __shared__ __align__(16) float    sWout[128];         // [2][64]
    __shared__ __align__(16) _Float16 sMask[NW * 3 * 1024];   // l1 | l1*t1 | l2*t2
    __shared__ __align__(16) _Float16 sStage[NW * 3 * 1024];  // bufA | bufR1 | bufR2
    __shared__ __align__(16) float    sX[NW * 16 * T_STEPS];

    const int tid = threadIdx.x;
    for (int i = tid; i < WSZ; i += blockDim.x) {
        sW[i]           = (_Float16)W_hh1[i];
        sW[WSZ + i]     = (_Float16)W_ih2[i];
        sW[2 * WSZ + i] = (_Float16)W_hh2[i];
    }
    for (int i = tid; i < 256; i += blockDim.x) {
        sB1[i]   = b_ih1[i] + b_hh1[i];
        sB2[i]   = b_ih2[i] + b_hh2[i];
        sWih1[i] = W_ih1[i];
    }
    if (tid < 128) sWout[tid] = W_out[tid];
    __syncthreads();

    const int wave = tid >> 5;
    const int lane = tid & 31;
    const int n    = lane & 15;
    const int hi   = lane >> 4;
    const long rowbase = (long)(blockIdx.x * NW + wave) * 16;

    _Float16* mL1 = sMask + wave * 3 * 1024;   // m_layer1
    _Float16* mP1 = mL1 + 1024;                // m_layer1*m_time1
    _Float16* mP2 = mL1 + 2048;                // m_layer2*m_time2
    _Float16* bufA  = sStage + wave * 3 * 1024;  // cell2 input (h1*l1), later raw h2
    _Float16* bufR1 = bufA + 1024;               // cell1 recurrent input
    _Float16* bufR2 = bufA + 2048;               // cell2 recurrent input
    float* xw = sX + wave * 16 * T_STEPS;

    // per-wave prologue: masks, initial recurrent inputs, x tile
    for (int i = lane; i < 1024; i += 32) {
        int r = i >> 6, c = i & 63;
        long g = (rowbase + r) * HDIM + c;
        float t1 = m_time1[g], t2 = m_time2[g];
        float L1 = m_layer1[g], L2 = m_layer2[g];
        mL1[i] = (_Float16)L1;
        mP1[i] = (_Float16)(L1 * t1);
        mP2[i] = (_Float16)(L2 * t2);
        bufR1[i] = (_Float16)(h1_0[g] * t1);   // first-step rec input (no layer mask)
        bufR2[i] = (_Float16)(h2_0[g] * t2);
    }
    for (int i = lane; i < 16 * T_STEPS; i += 32) {
        int r = i / T_STEPS, tt = i % T_STEPS;
        xw[i] = x[(rowbase + r) * T_STEPS + tt];
    }

    // c states + per-lane gate biases resident in VGPRs (D-layout: m=r+8*hi, col=g*16+n)
    float c1s[4][8], c2s[4][8];
    float bs1[4][4], bs2[4][4], wx1[4][4];
#pragma unroll
    for (int g = 0; g < 4; ++g) {
#pragma unroll
        for (int r = 0; r < 8; ++r) {
            int m = r + 8 * hi;
            long gi = (rowbase + m) * HDIM + g * 16 + n;
            c1s[g][r] = c1_0[gi];
            c2s[g][r] = c2_0[gi];
        }
#pragma unroll
        for (int q = 0; q < 4; ++q) {
            int col = q * 64 + g * 16 + n;
            bs1[g][q] = sB1[col];
            bs2[g][q] = sB2[col];
            wx1[g][q] = sWih1[col];
        }
    }
    lds_fence();

    for (int t = 0; t < T_STEPS; ++t) {
        // x values for this step (depend on r only; hoisted out of the g-loop)
        float xr[8];
#pragma unroll
        for (int r = 0; r < 8; ++r) {
            int m = r + 8 * hi;
            xr[r] = xw[m * T_STEPS + t];
        }

        // ---------------- cell 1: gates = h1rec @ W_hh1^T + x*W_ih1 + b ----------------
        v16h a0 = ldsA(bufR1, lane, 0);
        v16h a1 = ldsA(bufR1, lane, 32);
#pragma unroll
        for (int g = 0; g < 4; ++g) {
            v8f acc[4];
#pragma unroll
            for (int q = 0; q < 4; ++q) {
                int n0 = q * 64 + g * 16;
                v8f cc = {0.f, 0.f, 0.f, 0.f, 0.f, 0.f, 0.f, 0.f};
                cc = wmma(a0, ldsB(sW + (size_t)n0 * HDIM, lane, 0), cc);
                cc = wmma(a1, ldsB(sW + (size_t)n0 * HDIM, lane, 32), cc);
                acc[q] = cc;
            }
#pragma unroll
            for (int r = 0; r < 8; ++r) {
                int m = r + 8 * hi;
                float xv = xr[r];
                float ip = acc[0][r] + bs1[g][0] + xv * wx1[g][0];
                float fp = acc[1][r] + bs1[g][1] + xv * wx1[g][1];
                float gp = acc[2][r] + bs1[g][2] + xv * wx1[g][2];
                float op = acc[3][r] + bs1[g][3] + xv * wx1[g][3];
                float cn = sigf(fp) * c1s[g][r] + sigf(ip) * tanhfast(gp);
                float h  = sigf(op) * tanhfast(cn);
                c1s[g][r] = cn;
                int si = m * HDIM + g * 16 + n;
                bufA[si]  = (_Float16)(h * (float)mL1[si]);   // cell2 input this step
                bufR1[si] = (_Float16)(h * (float)mP1[si]);   // cell1 rec next step
            }
        }
        lds_fence();

        // -------- cell 2: gates = h1l1 @ W_ih2^T + h2rec @ W_hh2^T + b --------
        v16h u0 = ldsA(bufA, lane, 0);
        v16h u1 = ldsA(bufA, lane, 32);
        v16h w0 = ldsA(bufR2, lane, 0);
        v16h w1 = ldsA(bufR2, lane, 32);
#pragma unroll
        for (int g = 0; g < 4; ++g) {
            v8f acc[4];
#pragma unroll
            for (int q = 0; q < 4; ++q) {
                int n0 = q * 64 + g * 16;
                v8f cc = {0.f, 0.f, 0.f, 0.f, 0.f, 0.f, 0.f, 0.f};
                cc = wmma(u0, ldsB(sW + WSZ     + (size_t)n0 * HDIM, lane, 0),  cc);
                cc = wmma(u1, ldsB(sW + WSZ     + (size_t)n0 * HDIM, lane, 32), cc);
                cc = wmma(w0, ldsB(sW + 2 * WSZ + (size_t)n0 * HDIM, lane, 0),  cc);
                cc = wmma(w1, ldsB(sW + 2 * WSZ + (size_t)n0 * HDIM, lane, 32), cc);
                acc[q] = cc;
            }
#pragma unroll
            for (int r = 0; r < 8; ++r) {
                float ip = acc[0][r] + bs2[g][0];
                float fp = acc[1][r] + bs2[g][1];
                float gp = acc[2][r] + bs2[g][2];
                float op = acc[3][r] + bs2[g][3];
                float cn = sigf(fp) * c2s[g][r] + sigf(ip) * tanhfast(gp);
                float h  = sigf(op) * tanhfast(cn);
                c2s[g][r] = cn;
                int m = r + 8 * hi;
                int si = m * HDIM + g * 16 + n;
                bufR2[si] = (_Float16)(h * (float)mP2[si]);   // cell2 rec next step
                if (t == T_STEPS - 1) bufA[si] = (_Float16)h; // raw h2 for epilogue
            }
        }
        lds_fence();
    }

    // epilogue: out = (h2_raw * m_layer2) @ W_out^T + b_out
    if (lane < 16) {
        int row = lane;
        long gb = (rowbase + row) * HDIM;
        float s0 = 0.f, s1 = 0.f;
#pragma unroll
        for (int c = 0; c < HDIM; ++c) {
            float h = (float)bufA[row * HDIM + c] * m_layer2[gb + c];
            s0 += h * sWout[c];
            s1 += h * sWout[64 + c];
        }
        out[(rowbase + row) * 2 + 0] = s0 + b_out[0];
        out[(rowbase + row) * 2 + 1] = s1 + b_out[1];
    }
}

extern "C" void kernel_launch(void* const* d_in, const int* in_sizes, int n_in,
                              void* d_out, int out_size, void* d_ws, size_t ws_size,
                              hipStream_t stream) {
    const float* x      = (const float*)d_in[0];
    const float* W_ih1  = (const float*)d_in[1];
    const float* W_hh1  = (const float*)d_in[2];
    const float* b_ih1  = (const float*)d_in[3];
    const float* b_hh1  = (const float*)d_in[4];
    const float* W_ih2  = (const float*)d_in[5];
    const float* W_hh2  = (const float*)d_in[6];
    const float* b_ih2  = (const float*)d_in[7];
    const float* b_hh2  = (const float*)d_in[8];
    const float* W_out  = (const float*)d_in[9];
    const float* b_out  = (const float*)d_in[10];
    const float* h1_0   = (const float*)d_in[11];
    const float* c1_0   = (const float*)d_in[12];
    const float* h2_0   = (const float*)d_in[13];
    const float* c2_0   = (const float*)d_in[14];
    const float* m_t1   = (const float*)d_in[15];
    const float* m_t2   = (const float*)d_in[16];
    const float* m_l1   = (const float*)d_in[17];
    const float* m_l2   = (const float*)d_in[18];
    float* out = (float*)d_out;

    int B = in_sizes[11] / HDIM;            // h1_0 is [B, H]
    int rows_per_block = NW * 16;           // 128
    int grid = (B + rows_per_block - 1) / rows_per_block;

    hipLaunchKernelGGL(vlstm_kernel, dim3(grid), dim3(NW * 32), 0, stream,
                       x, W_ih1, W_hh1, b_ih1, b_hh1, W_ih2, W_hh2, b_ih2, b_hh2,
                       W_out, b_out, h1_0, c1_0, h2_0, c2_0,
                       m_t1, m_t2, m_l1, m_l2, out);
}
